// SoftVectorQuantizer_61220463837255
// MI455X (gfx1250) — compile-verified, matrix operand order checked by reading, not run on hardware
//
#include <hip/hip_runtime.h>

// ---------------------------------------------------------------------------
// Soft vector quantizer, fused: logits GEMM + dual-temperature softmax +
// probs GEMM + diagnostics.  bf16 WMMA (v_wmma_f32_16x16x32_bf16) on gfx1250.
// Double-buffered async-to-LDS pipeline; pass-2 uses a pre-transposed
// codebook tile (dim-major, padded rows) so BOTH gemms load fragments with
// contiguous ds_load_b128.  Per-lane online softmax stats.
// ---------------------------------------------------------------------------

typedef __attribute__((ext_vector_type(16))) __bf16 v16bf;
typedef __attribute__((ext_vector_type(8)))  float  v8f;
typedef __attribute__((ext_vector_type(4)))  int    i32x4;
typedef __attribute__((address_space(1)))    i32x4  gas_i32x4;  // global
typedef __attribute__((address_space(3)))    i32x4  las_i32x4;  // LDS

#define MTOK   8192      // tokens (8*1024)
#define NCODE  8192      // codebook entries
#define KDIM   256       // embedding dim
#define BM     64        // token rows per workgroup
#define BN     32        // codes per block iteration
#define NBLK   (NCODE / BN)
#define ETS    40        // Et row stride in codes (80 B, bank-friendly padding)
#define INV_TAU (1.0f/0.07f)
#define INV_TE  (100.0f)         // 1/0.01 entropy temperature
#define ENT_OFF  2097152
#define IDX_OFF  2097153
#define AVG_OFF  2105345
#define MAXP_OFF 2105346
#define COS_OFF  2105347
#define EMBBF_OFF 8704           // float offset in ws of bf16 codebook
#define EMBT_OFF  1057280        // float offset in ws of transposed bf16 codebook

#if __has_builtin(__builtin_amdgcn_global_load_async_to_lds_b128) && \
    __has_builtin(__builtin_amdgcn_s_wait_asynccnt)
#define USE_ASYNC 1
#else
#define USE_ASYNC 0
#endif

__device__ __forceinline__ float redsum16(float v) {
  #pragma unroll
  for (int m = 1; m < 16; m <<= 1) v += __shfl_xor(v, m, 32);
  return v;
}

// log-sum-exp merges (associative online-softmax combine)
__device__ __forceinline__ void merge_lse(float& m, float& s, float om, float os) {
  const float mn = fmaxf(m, om);
  s = s * __expf(m - mn) + os * __expf(om - mn);
  m = mn;
}
__device__ __forceinline__ void merge_lsew(float& m, float& s, float& w,
                                           float om, float os, float ow) {
  const float mn = fmaxf(m, om);
  const float dA = m - mn, dB = om - mn;
  const float eA = __expf(dA), eB = __expf(dB);
  w = eA * (w + dA * s) + eB * (ow + dB * os);
  s = eA * s + eB * os;
  m = mn;
}

__global__ void svq_init(float* ws) {
  int i = blockIdx.x * blockDim.x + threadIdx.x;
  if (i < 4 + NCODE) ws[i] = 0.0f;
}

// one-time f32 -> bf16 codebook conversion (4 elems / thread)
__global__ void svq_cvt(const float* __restrict__ src, __bf16* __restrict__ dst) {
  const int i = blockIdx.x * blockDim.x + threadIdx.x;
  const float4 v = ((const float4*)src)[i];
  __bf16* d = dst + (size_t)i * 4;
  d[0] = (__bf16)v.x; d[1] = (__bf16)v.y; d[2] = (__bf16)v.z; d[3] = (__bf16)v.w;
}

// one-time transposed bf16 codebook: dst[d][c] = src[c][d]
__global__ void svq_cvtT(const float* __restrict__ src, __bf16* __restrict__ dst) {
  const int idx = blockIdx.x * blockDim.x + threadIdx.x;
  const int d  = idx >> 11;          // 0..255
  const int c4 = (idx & 2047) * 4;   // code base
  __bf16* out = dst + (size_t)d * NCODE + c4;
  #pragma unroll
  for (int i = 0; i < 4; ++i) out[i] = (__bf16)src[(size_t)(c4 + i) * KDIM + d];
}

// async stage of a BN x KDIM bf16 tile into LDS (row-major; 64 B per thread)
__device__ __forceinline__ void stage_issue(const __bf16* __restrict__ embbf, int nb,
                                            __bf16* dst_base, int tid) {
  const __bf16* src = embbf + (size_t)nb * BN * KDIM + tid * 32;
  __bf16* dst = dst_base + tid * 32;
#if USE_ASYNC
  #pragma unroll
  for (int i = 0; i < 4; ++i)
    __builtin_amdgcn_global_load_async_to_lds_b128(
        (gas_i32x4*)(src + i * 8), (las_i32x4*)(dst + i * 8), 0, 0);
#else
  const uint4* s4 = (const uint4*)src;
  uint4* d4 = (uint4*)dst;
  #pragma unroll
  for (int i = 0; i < 4; ++i) d4[i] = s4[i];
#endif
}
// async stage of a KDIM x BN dim-major tile (thread t = dim t, padded rows)
__device__ __forceinline__ void stage_issue_et(const __bf16* __restrict__ embT, int nb,
                                               __bf16* dst_base, int tid) {
  const __bf16* src = embT + (size_t)tid * NCODE + nb * BN;
  __bf16* dst = dst_base + tid * ETS;
#if USE_ASYNC
  #pragma unroll
  for (int i = 0; i < 4; ++i)
    __builtin_amdgcn_global_load_async_to_lds_b128(
        (gas_i32x4*)(src + i * 8), (las_i32x4*)(dst + i * 8), 0, 0);
#else
  const uint4* s4 = (const uint4*)src;
  #pragma unroll
  for (int i = 0; i < 4; ++i) ((uint4*)dst)[i] = s4[i];
#endif
}
__device__ __forceinline__ void stage_wait() {
#if USE_ASYNC
  __builtin_amdgcn_s_wait_asynccnt(0);
#endif
}

__global__ __launch_bounds__(256, 1)
void svq_main(const float* __restrict__ z, const __bf16* __restrict__ embbf,
              const __bf16* __restrict__ embT,
              float* __restrict__ zq_out, float* __restrict__ idx_out,
              float* __restrict__ ws_scal, float* __restrict__ colsum) {
  __shared__ __bf16 ldsE[2][BN * KDIM];  // 2 x 16 KB row-major codebook tiles
  __shared__ __bf16 ldsEt[KDIM * ETS];   // 20 KB dim-major tile (pass 2)
  __shared__ __bf16 ldsP[BM * BN];       // 4 KB  probs staging (C->A transpose)
  __shared__ float  stats[4][2][16][8];  // 4 KB  per-row softmax stats
  __shared__ float  rowred[BM][4];       // 1 KB  cosine reduction

  const int tid  = threadIdx.x;
  const int lane = tid & 31;
  const int wave = tid >> 5;
  const int wm   = wave >> 1;     // 0..3: 16-row strip
  const int wn   = wave & 1;      // 0..1: column half
  const int h    = lane >> 4;     // lane half (K-chunk select)
  const int ln   = lane & 15;
  const int r0   = blockIdx.x * BM;

  // ---- A fragments: this wave's 16 token rows, all K=256, bf16, resident ----
  v16bf afrag[8];
  {
    const float* zr = z + (size_t)(r0 + wm * 16 + ln) * KDIM;
    #pragma unroll
    for (int ks = 0; ks < 8; ++ks) {
      const int kb = ks * 32 + h * 8;
      #pragma unroll
      for (int j = 0; j < 8; ++j) {
        afrag[ks][j]     = (__bf16)zr[kb + j];
        afrag[ks][8 + j] = (__bf16)zr[kb + 16 + j];
      }
    }
  }

  // per-LANE online stats for rows (r + 8*h)
  float m1[8], s1[8], m2[8], s2[8], w2[8], av[8]; int ai[8];
  #pragma unroll
  for (int r = 0; r < 8; ++r) {
    m1[r] = -1e30f; s1[r] = 0.f; m2[r] = -1e30f; s2[r] = 0.f; w2[r] = 0.f;
    av[r] = -1e30f; ai[r] = 0;
  }

  // ========================= PASS 1: softmax stats =========================
  stage_issue(embbf, 0, ldsE[0], tid);
  for (int nb = 0; nb < NBLK; ++nb) {
    stage_wait();
    __syncthreads();
    if (nb + 1 < NBLK) stage_issue(embbf, nb + 1, ldsE[(nb + 1) & 1], tid);
    const __bf16* E = ldsE[nb & 1];

    v8f acc = {};
    #pragma unroll
    for (int ks = 0; ks < 8; ++ks) {
      v16bf b;
      const __bf16* eb = &E[(size_t)(wn * 16 + ln) * KDIM + ks * 32 + h * 8];
      #pragma unroll
      for (int j = 0; j < 8; ++j) { b[j] = eb[j]; b[8 + j] = eb[16 + j]; }
      acc = __builtin_amdgcn_wmma_f32_16x16x32_bf16(false, afrag[ks], false, b,
                                                    (short)0, acc, false, false);
    }
    const int code = nb * BN + wn * 16 + ln;
    #pragma unroll
    for (int r = 0; r < 8; ++r) {
      const float l = acc[r];
      if (l > av[r]) { av[r] = l; ai[r] = code; }
      const float u1 = l * INV_TAU;
      {
        const float mn = fmaxf(m1[r], u1);
        s1[r] = s1[r] * __expf(m1[r] - mn) + __expf(u1 - mn);
        m1[r] = mn;
      }
      const float u2 = l * INV_TE;
      {
        const float mn = fmaxf(m2[r], u2);
        const float d = m2[r] - mn, t = u2 - mn;
        const float sc = __expf(d), ex = __expf(t);
        w2[r] = sc * (w2[r] + d * s2[r]) + ex * t;
        s2[r] = sc * s2[r] + ex;
        m2[r] = mn;
      }
    }
  }

  // prefetch pass-2 tile 0; it transfers while we merge stats
  stage_issue(embbf, 0, ldsE[0], tid);

  // ---- single cross-lane merge over the 16 columns of each lane group ----
  #pragma unroll
  for (int r = 0; r < 8; ++r) {
    #pragma unroll
    for (int msk = 1; msk < 16; msk <<= 1) {
      const float om  = __shfl_xor(m1[r], msk, 32);
      const float os  = __shfl_xor(s1[r], msk, 32);
      merge_lse(m1[r], s1[r], om, os);
      const float om2 = __shfl_xor(m2[r], msk, 32);
      const float os2 = __shfl_xor(s2[r], msk, 32);
      const float ow2 = __shfl_xor(w2[r], msk, 32);
      merge_lsew(m2[r], s2[r], w2[r], om2, os2, ow2);
      const float oav = __shfl_xor(av[r], msk, 32);
      const int   oai = __shfl_xor(ai[r], msk, 32);
      if (oav > av[r] || (oav == av[r] && oai < ai[r])) { av[r] = oav; ai[r] = oai; }
    }
  }

  // ---- merge stats across the wn pair via LDS ----
  if (ln == 0) {
    #pragma unroll
    for (int r = 0; r < 8; ++r) {
      float* st = stats[wm][wn][r + 8 * h];
      st[0] = m1[r]; st[1] = s1[r]; st[2] = m2[r]; st[3] = s2[r];
      st[4] = w2[r]; st[5] = av[r]; st[6] = (float)ai[r];
    }
  }
  __syncthreads();
  float fm1[8], fi1[8], fm2[8], fi2[8]; int fai[8];
  #pragma unroll
  for (int r = 0; r < 8; ++r) {
    const float* a0 = stats[wm][0][r + 8 * h];
    const float* a1 = stats[wm][1][r + 8 * h];
    float mm = a0[0], ss = a0[1];
    merge_lse(mm, ss, a1[0], a1[1]);
    float mm2 = a0[2], ss2 = a0[3], ww2 = a0[4];
    merge_lsew(mm2, ss2, ww2, a1[2], a1[3], a1[4]);
    fm1[r] = mm;  fi1[r] = 1.0f / ss;
    fm2[r] = mm2; fi2[r] = 1.0f / ss2;
    const float v0 = a0[5], v1 = a1[5];
    const int i0 = (int)a0[6], i1 = (int)a1[6];
    fai[r] = (v1 > v0 || (v1 == v0 && i1 < i0)) ? i1 : i0;

    if (wn == 0 && ln == 0) {
      const int row = r0 + wm * 16 + r + 8 * h;
      idx_out[row] = (float)fai[r];
      atomicAdd(&ws_scal[0], fi1[r]);                        // p_max = 1/s1
      atomicAdd(&ws_scal[1], -(ww2 * fi2[r] - __logf(ss2))); // -sum p*log p
    }
  }

  // ================= PASS 2: exact probs, z_q, column sums =================
  const v8f vzero = {};
  v8f zacc[8];
  #pragma unroll
  for (int ct = 0; ct < 8; ++ct) zacc[ct] = vzero;

  for (int nb = 0; nb < NBLK; ++nb) {
    stage_wait();
    __syncthreads();
    if (nb + 1 < NBLK) stage_issue(embbf, nb + 1, ldsE[(nb + 1) & 1], tid);
    stage_issue_et(embT, nb, ldsEt, tid);   // dim-major tile, used after mid wait
    const __bf16* E = ldsE[nb & 1];

    // logits tile (recompute)
    v8f acc = {};
    #pragma unroll
    for (int ks = 0; ks < 8; ++ks) {
      v16bf b;
      const __bf16* eb = &E[(size_t)(wn * 16 + ln) * KDIM + ks * 32 + h * 8];
      #pragma unroll
      for (int j = 0; j < 8; ++j) { b[j] = eb[j]; b[8 + j] = eb[16 + j]; }
      acc = __builtin_amdgcn_wmma_f32_16x16x32_bf16(false, afrag[ks], false, b,
                                                    (short)0, acc, false, false);
    }
    const int code = nb * BN + wn * 16 + ln;
    float csum = 0.f;
    #pragma unroll
    for (int r = 0; r < 8; ++r) {
      const float l = acc[r];
      const float p1 = __expf(l * INV_TAU - fm1[r]) * fi1[r];
      const float p2 = __expf(l * INV_TE  - fm2[r]) * fi2[r];
      csum += p2;
      ldsP[(size_t)(wm * 16 + r + 8 * h) * BN + (wn * 16 + ln)] = (__bf16)p1;
    }
    csum += __shfl_xor(csum, 16, 32);
    if (h == 0) atomicAdd(&colsum[code], csum);

    stage_wait();        // Et (and next E) transfers complete
    __syncthreads();     // ldsP + ldsEt ready for everyone

    // ---- z_q += probs_blk @ E_blk: both operands contiguous in LDS ----
    v16bf ap;
    {
      const __bf16* pr = &ldsP[(size_t)(wm * 16 + ln) * BN + h * 8];
      #pragma unroll
      for (int j = 0; j < 8; ++j) { ap[j] = pr[j]; ap[8 + j] = pr[16 + j]; }
    }
    #pragma unroll
    for (int ct = 0; ct < 8; ++ct) {
      const int dim = wn * 128 + ct * 16 + ln;
      v16bf b;
      const __bf16* eb = &ldsEt[(size_t)dim * ETS + h * 8];
      #pragma unroll
      for (int j = 0; j < 8; ++j) { b[j] = eb[j]; b[8 + j] = eb[16 + j]; }
      zacc[ct] = __builtin_amdgcn_wmma_f32_16x16x32_bf16(false, ap, false, b,
                                                         (short)0, zacc[ct], false, false);
    }
  }

  // ---- z_q store + cosine diagnostics ----
  __syncthreads();
  for (int i = tid; i < BM * 4; i += 256) ((float*)rowred)[i] = 0.f;
  __syncthreads();

  float dots[8], nqs[8], nzs[8];
  #pragma unroll
  for (int r = 0; r < 8; ++r) { dots[r] = 0.f; nqs[r] = 0.f; nzs[r] = 0.f; }
  #pragma unroll
  for (int ct = 0; ct < 8; ++ct) {
    const int dim = wn * 128 + ct * 16 + ln;
    #pragma unroll
    for (int r = 0; r < 8; ++r) {
      const int row = r0 + wm * 16 + r + 8 * h;
      const float q = zacc[ct][r];
      zq_out[(size_t)row * KDIM + dim] = q;
      const float zv = z[(size_t)row * KDIM + dim];
      dots[r] += q * zv; nqs[r] += q * q; nzs[r] += zv * zv;
    }
  }
  #pragma unroll
  for (int r = 0; r < 8; ++r) {
    const float d  = redsum16(dots[r]);
    const float nq = redsum16(nqs[r]);
    const float nz = redsum16(nzs[r]);
    if (ln == 0) {
      const int rl = wm * 16 + r + 8 * h;
      atomicAdd(&rowred[rl][0], d);
      atomicAdd(&rowred[rl][1], nq);
      atomicAdd(&rowred[rl][2], nz);
    }
  }
  __syncthreads();
  if (tid < BM) {
    const float d = rowred[tid][0], nq = rowred[tid][1], nz = rowred[tid][2];
    const float c = d / (fmaxf(sqrtf(nz), 1e-8f) * fmaxf(sqrtf(nq), 1e-8f));
    atomicAdd(&ws_scal[2], c);
  }
}

__global__ void svq_final(const float* __restrict__ ws_scal,
                          const float* __restrict__ colsum,
                          float* __restrict__ out) {
  __shared__ float red[256];
  float loc = 0.f;
  for (int k = threadIdx.x; k < NCODE; k += 256) {
    const float a = colsum[k] * (1.0f / MTOK);
    loc += -a * __logf(a + 1e-6f);
  }
  red[threadIdx.x] = loc;
  __syncthreads();
  for (int s = 128; s > 0; s >>= 1) {
    if ((int)threadIdx.x < s) red[threadIdx.x] += red[threadIdx.x + s];
    __syncthreads();
  }
  if (threadIdx.x == 0) {
    const float avg_entropy = red[0];
    const float sample_entropy = ws_scal[1] * (1.0f / MTOK);
    out[ENT_OFF]  = 0.01f * (sample_entropy - avg_entropy);
    out[AVG_OFF]  = 1.0f / (float)NCODE;
    out[MAXP_OFF] = ws_scal[0] * (1.0f / MTOK);
    out[COS_OFF]  = ws_scal[2] * (1.0f / MTOK);
  }
}

extern "C" void kernel_launch(void* const* d_in, const int* in_sizes, int n_in,
                              void* d_out, int out_size, void* d_ws, size_t ws_size,
                              hipStream_t stream) {
  const float* z   = (const float*)d_in[0];   // [8,1024,256] f32
  const float* emb = (const float*)d_in[1];   // [1,8192,256] f32
  float* out = (float*)d_out;
  float* ws  = (float*)d_ws;   // [0..3] scalars, [4..) colsums, bf16 emb, bf16 embT
  __bf16* embbf = (__bf16*)(ws + EMBBF_OFF);
  __bf16* embT  = (__bf16*)(ws + EMBT_OFF);

  svq_init<<<(4 + NCODE + 255) / 256, 256, 0, stream>>>(ws);
  svq_cvt<<<(NCODE * KDIM / 4) / 256, 256, 0, stream>>>(emb, embbf);
  svq_cvtT<<<(KDIM * NCODE / 4) / 256, 256, 0, stream>>>(emb, embT);
  svq_main<<<MTOK / BM, 256, 0, stream>>>(z, embbf, embT,
                                          out /* z_q at offset 0 */,
                                          out + IDX_OFF, ws, ws + 4);
  svq_final<<<1, 256, 0, stream>>>(ws, ws + 4, out);
}
